// QResnetBlock_46325517254886
// MI455X (gfx1250) — matrix operand order repaired and checked by reading, not compiled, not run on hardware
//
#include <hip/hip_runtime.h>
#include <math.h>

typedef __attribute__((ext_vector_type(16))) _Float16 v16h;
typedef __attribute__((ext_vector_type(8)))  _Float16 v8h;
typedef __attribute__((ext_vector_type(8)))  float    v8f;

#define QC 13
#define NQ 12
#define B_SZ 128
#define DIM 120
#define CONV_C 81
#define NS 4096          // 2^12 amplitudes
#define PI_F 3.14159265358979323846f

__device__ __forceinline__ float swishf(float v) {
  // x * sigmoid(x) with v_rcp_f32 instead of IEEE divide
  return v * __builtin_amdgcn_rcpf(1.0f + __expf(-v));
}

// ---------------------------------------------------------------------------
// Kernel 1: quantum stage. One block (128 thr = 4 waves) per (batch, channel)
// circuit. 12-qubit state (4096 cplx = 32KB) lives entirely in LDS.
// CRZ quads are fused into a single elementwise pass (diagonal phases).
// ---------------------------------------------------------------------------
__global__ __launch_bounds__(128) void quantum_kernel(
    const float* __restrict__ x, const float* __restrict__ qparams,
    float* __restrict__ h1) {
  __shared__ float2 psi[NS];          // 32 KB state vector
  __shared__ float cg_s[72], sg_s[72];
  __shared__ float czw[NQ], szw[NQ];
  __shared__ float red[128 * NQ];

  const int b   = blockIdx.x;   // batch
  const int qc  = blockIdx.y;   // quantum channel
  const int tid = threadIdx.x;  // 0..127

  // per-gate half-angle cos/sin: ang = 2*pi*theta -> half-angle = pi*theta
  if (tid < 72) {
    float th = qparams[qc * 72 + tid];
    float ha = PI_F * th;
    cg_s[tid] = cosf(ha);
    sg_s[tid] = sinf(ha);
  }
  // embedding angles: z[w] = x[b, pos, 3*qc + j], w = pos*3 + j
  if (tid < NQ) {
    int pos = tid / 3, j = tid % 3;
    float z = x[(b * 4 + pos) * DIM + 3 * qc + j];
    czw[tid] = cosf(0.5f * z);
    szw[tid] = sinf(0.5f * z);
  }
  __syncthreads();

  // init product state: psi[s] = (-i)^popcount(s) * prod_w (bit? sin : cos)
  for (int i = 0; i < 32; ++i) {
    int s = tid + (i << 7);
    float mag = 1.0f;
#pragma unroll
    for (int w = 0; w < NQ; ++w)
      mag *= ((s >> (11 - w)) & 1) ? szw[w] : czw[w];
    int k = __popc((unsigned)s) & 3;
    float2 v;
    v.x = (k == 0) ? mag : (k == 2) ? -mag : 0.0f;
    v.y = (k == 1) ? -mag : (k == 3) ? mag : 0.0f;
    psi[s] = v;
  }

  // 3 layers x 6 groups: even group = fused CRZ quad, odd group = 4 CRX gates
  // control/target bits: bc = {8,0,4}[blk] + i4 ; bt = {0,4,8}[blk] + i4
  for (int l = 0; l < 3; ++l) {
    for (int q = 0; q < 6; ++q) {
      const int blk = q >> 1;
      const int gb  = l * 24 + q * 4;            // first theta index of quad
      const int bcb = (blk == 0) ? 8 : (blk == 1) ? 0 : 4;
      const int btb = (blk == 0) ? 0 : (blk == 1) ? 4 : 8;
      if ((q & 1) == 0) {
        // ---- fused CRZ quad: one pass, product of <=4 diagonal phases ----
        __syncthreads();
        float c0 = cg_s[gb + 0], s0 = sg_s[gb + 0];
        float c1 = cg_s[gb + 1], s1 = sg_s[gb + 1];
        float c2 = cg_s[gb + 2], s2 = sg_s[gb + 2];
        float c3 = cg_s[gb + 3], s3 = sg_s[gb + 3];
        for (int i = 0; i < 32; ++i) {
          int s = tid + (i << 7);
          float pr = 1.0f, pi = 0.0f;
#pragma unroll
          for (int j = 0; j < 4; ++j) {
            if ((s >> (bcb + j)) & 1) {
              float cg = (j == 0) ? c0 : (j == 1) ? c1 : (j == 2) ? c2 : c3;
              float sg = (j == 0) ? s0 : (j == 1) ? s1 : (j == 2) ? s2 : s3;
              float ps = ((s >> (btb + j)) & 1) ? sg : -sg;
              float nr = pr * cg - pi * ps;
              pi = pr * ps + pi * cg;
              pr = nr;
            }
          }
          float2 v = psi[s];
          float2 r;
          r.x = v.x * pr - v.y * pi;
          r.y = v.x * pi + v.y * pr;
          psi[s] = r;
        }
      } else {
        // ---- CRX quad: 4 sequential pair-mixing gates ----
        for (int j = 0; j < 4; ++j) {
          const int bc = bcb + j, bt = btb + j;
          const float cg = cg_s[gb + j], sg = sg_s[gb + j];
          const int b0 = bc < bt ? bc : bt;
          const int b1 = bc < bt ? bt : bc;
          const unsigned m0 = (1u << b0) - 1u, m1 = (1u << b1) - 1u;
          __syncthreads();
          for (int i = 0; i < 8; ++i) {
            unsigned p  = (unsigned)(tid + (i << 7));          // 0..1023
            unsigned u  = ((p & ~m0) << 1) | (p & m0);          // insert 0 @ b0
            unsigned s0i = ((u & ~m1) << 1) | (u & m1);         // insert 0 @ b1
            s0i |= (1u << bc);                                  // control = 1
            unsigned s1i = s0i | (1u << bt);
            float2 a0 = psi[s0i], a1 = psi[s1i];
            float2 r0, r1;
            r0.x = cg * a0.x + sg * a1.y;  r0.y = cg * a0.y - sg * a1.x;
            r1.x = cg * a1.x + sg * a0.y;  r1.y = cg * a1.y - sg * a0.x;
            psi[s0i] = r0;
            psi[s1i] = r1;
          }
        }
      }
    }
  }
  __syncthreads();

  // PauliZ expectations: E[w] = sum_s |psi_s|^2 * (1 - 2*bit_{11-w}(s))
  float acc[NQ];
#pragma unroll
  for (int w = 0; w < NQ; ++w) acc[w] = 0.0f;
  for (int i = 0; i < 32; ++i) {
    int s = tid + (i << 7);
    float2 v = psi[s];
    float pr = v.x * v.x + v.y * v.y;
#pragma unroll
    for (int w = 0; w < NQ; ++w)
      acc[w] += ((s >> (11 - w)) & 1) ? -pr : pr;
  }
#pragma unroll
  for (int w = 0; w < NQ; ++w) red[tid * NQ + w] = acc[w];
  __syncthreads();
  for (int off = 64; off >= 1; off >>= 1) {
    if (tid < off)
#pragma unroll
      for (int w = 0; w < NQ; ++w)
        red[tid * NQ + w] += red[(tid + off) * NQ + w];
    __syncthreads();
  }
  if (tid < NQ) {
    int pos = tid / 3, j = tid % 3;
    h1[(b * 4 + pos) * (3 * QC) + qc * 3 + j] = red[tid];
  }
}

// ---------------------------------------------------------------------------
// Kernel 2: weight standardization, one block per (conv, out-channel).
// ---------------------------------------------------------------------------
__global__ __launch_bounds__(256) void wstd_kernel(
    const float* __restrict__ k0, const float* __restrict__ k1,
    float* __restrict__ o0, float* __restrict__ o1) {
  __shared__ float s1[256], s2[256];
  const int which = blockIdx.x / CONV_C;
  const int oc    = blockIdx.x % CONV_C;
  const float* src = which ? k1 : k0;
  float*       dst = which ? o1 : o0;
  const int tid = threadIdx.x;
  float a = 0.0f, q = 0.0f;
  for (int t = tid; t < 729; t += 256) {
    float v = src[t * CONV_C + oc];
    a += v; q += v * v;
  }
  s1[tid] = a; s2[tid] = q;
  __syncthreads();
  for (int off = 128; off >= 1; off >>= 1) {
    if (tid < off) { s1[tid] += s1[tid + off]; s2[tid] += s2[tid + off]; }
    __syncthreads();
  }
  float mean = s1[0] * (1.0f / 729.0f);
  float var  = s2[0] * (1.0f / 729.0f) - mean * mean;
  float istd = rsqrtf(var + 1e-5f);
  for (int t = tid; t < 729; t += 256)
    dst[t * CONV_C + oc] = (src[t * CONV_C + oc] - mean) * istd;
}

// ---------------------------------------------------------------------------
// Kernel 3a: prep for the WMMA GEMM: swish(temb) -> f16, and td_w -> f16
// transposed to [N][K] so B fragments are pairs of 16B (b128) loads.
// ---------------------------------------------------------------------------
__global__ __launch_bounds__(256) void prep_mlp_kernel(
    const float* __restrict__ temb, const float* __restrict__ tdw,
    _Float16* __restrict__ swA, _Float16* __restrict__ bT) {
  const int i = blockIdx.x * 256 + threadIdx.x;     // 0 .. 122879
  if (i < B_SZ * 512) {
    float v = temb[i];
    swA[i] = (_Float16)swishf(v);
  }
  // tdw is (512 x 240) row-major; bT is (240 x 512)
  int k = i / 240, n = i % 240;
  bT[n * 512 + k] = (_Float16)tdw[i];
}

// ---------------------------------------------------------------------------
// Kernel 3b: time MLP GEMM via WMMA, fully unrolled: per 16x16 tile,
// 16 k-steps of 32 -> 16 v_wmma_f32_16x16x32_f16 + pure b128 loads.
// ---------------------------------------------------------------------------
__global__ __launch_bounds__(32) void time_mlp_wmma(
    const _Float16* __restrict__ swA, const _Float16* __restrict__ bT,
    const float* __restrict__ bias, float* __restrict__ out) {
  const int nt   = blockIdx.x;            // 0..14 (N tiles of 240)
  const int mt   = blockIdx.y;            // 0..7  (M tiles of 128)
  const int lane = threadIdx.x;           // wave32
  const int half = lane >> 4;
  const int ml   = lane & 15;

  // A 16x32 f16 layout: lane<16 -> M=lane, e<8: K=e+half*8, e>=8: K=e+8+half*8
  const _Float16* ap = swA + (mt * 16 + ml) * 512 + half * 8;
  // B 32x16 f16 layout: lane<16 -> N=lane, e<8: K=e+half*8, e>=8: K=e+8+half*8
  const _Float16* bp = bT + (nt * 16 + ml) * 512 + half * 8;

  v8f c = {};
#pragma unroll
  for (int k0 = 0; k0 < 512; k0 += 32) {
    v8h alo = *(const v8h*)(ap + k0);
    v8h ahi = *(const v8h*)(ap + k0 + 16);
    v16h a  = __builtin_shufflevector(alo, ahi, 0, 1, 2, 3, 4, 5, 6, 7,
                                      8, 9, 10, 11, 12, 13, 14, 15);
    v8h blo = *(const v8h*)(bp + k0);
    v8h bhi = *(const v8h*)(bp + k0 + 16);
    v16h bfr = __builtin_shufflevector(blo, bhi, 0, 1, 2, 3, 4, 5, 6, 7,
                                       8, 9, 10, 11, 12, 13, 14, 15);
    c = __builtin_amdgcn_wmma_f32_16x16x32_f16(false, a, false, bfr,
                                               (short)0, c, false, false);
  }
  const int col = nt * 16 + ml;
#pragma unroll
  for (int r = 0; r < 8; ++r) {
    int orow = mt * 16 + half * 8 + r;    // C/D layout: VGPR r, lane half
    out[orow * 240 + col] = c[r] + bias[col];
  }
}

// ---------------------------------------------------------------------------
// Kernel 4: fused conv0 -> concat -> GN0 -> time-mod -> swish -> conv1 ->
// concat(h1 stage-1, the source bug) -> GN1 -> swish -> residual.
// One block per batch sample; H=W=2 so everything is sample-local.
// ---------------------------------------------------------------------------
__global__ __launch_bounds__(256) void fused_block_kernel(
    const float* __restrict__ x,   const float* __restrict__ h1,
    const float* __restrict__ tml, const float* __restrict__ k0s,
    const float* __restrict__ b0,  const float* __restrict__ k1s,
    const float* __restrict__ b1,  const float* __restrict__ g0s,
    const float* __restrict__ g0b, const float* __restrict__ g1s,
    const float* __restrict__ g1b, float* __restrict__ out) {
  __shared__ float xs[480], hq[156], hA[480], hB[480], tm[240];
  __shared__ float gmu[10], gis[10];
  const int bI = blockIdx.x, tid = threadIdx.x;

  // warm L2 for conv1 weights while conv0 runs (global_prefetch_b8)
  __builtin_prefetch(k1s + tid * 231, 0, 0);

  for (int i = tid; i < 480; i += 256) xs[i] = x[bI * 480 + i];
  for (int i = tid; i < 156; i += 256) hq[i] = h1[bI * 156 + i];
  for (int i = tid; i < 240; i += 256) tm[i] = tml[bI * 240 + i];
  __syncthreads();

  // conv0 on x[...,39:] (+bias), concat with quantum h1 -> hA
  for (int o = tid; o < 480; o += 256) {
    int pos = o / DIM, ch = o % DIM;
    if (ch < 39) { hA[o] = hq[pos * 39 + ch]; continue; }
    int oc = ch - 39;
    int ph = pos >> 1, pw = pos & 1;
    float acc = b0[oc];
    for (int dh = 0; dh < 3; ++dh) {
      int ih = ph + dh - 1; if ((unsigned)ih > 1u) continue;
      for (int dw = 0; dw < 3; ++dw) {
        int iw = pw + dw - 1; if ((unsigned)iw > 1u) continue;
        const float* kk  = k0s + (dh * 3 + dw) * 6561 + oc;
        const float* xin = xs + (ih * 2 + iw) * DIM + 39;
        for (int ic = 0; ic < CONV_C; ++ic) acc += xin[ic] * kk[ic * CONV_C];
      }
    }
    hA[o] = acc;
  }
  __syncthreads();

  // GN0 stats: 10 groups x (4 pos x 12 ch)
  if (tid < 10) {
    float s = 0.0f, q = 0.0f;
    for (int pos = 0; pos < 4; ++pos)
      for (int j = 0; j < 12; ++j) {
        float v = hA[pos * DIM + tid * 12 + j];
        s += v; q += v * v;
      }
    float mu = s * (1.0f / 48.0f);
    gmu[tid] = mu;
    gis[tid] = rsqrtf(q * (1.0f / 48.0f) - mu * mu + 1e-6f);
  }
  __syncthreads();

  // normalize + time scale/shift + swish (in place)
  for (int o = tid; o < 480; o += 256) {
    int ch = o % DIM, g = ch / 12;
    float v = (hA[o] - gmu[g]) * gis[g] * g0s[ch] + g0b[ch];
    v = v * (1.0f + tm[ch]) + tm[120 + ch];
    hA[o] = swishf(v);
  }
  __syncthreads();

  // conv1 on modulated h[...,39:], concat with STAGE-1 h1 (source bug) -> hB
  for (int o = tid; o < 480; o += 256) {
    int pos = o / DIM, ch = o % DIM;
    if (ch < 39) { hB[o] = hq[pos * 39 + ch]; continue; }
    int oc = ch - 39;
    int ph = pos >> 1, pw = pos & 1;
    float acc = b1[oc];
    for (int dh = 0; dh < 3; ++dh) {
      int ih = ph + dh - 1; if ((unsigned)ih > 1u) continue;
      for (int dw = 0; dw < 3; ++dw) {
        int iw = pw + dw - 1; if ((unsigned)iw > 1u) continue;
        const float* kk  = k1s + (dh * 3 + dw) * 6561 + oc;
        const float* xin = hA + (ih * 2 + iw) * DIM + 39;
        for (int ic = 0; ic < CONV_C; ++ic) acc += xin[ic] * kk[ic * CONV_C];
      }
    }
    hB[o] = acc;
  }
  __syncthreads();

  // GN1 stats
  if (tid < 10) {
    float s = 0.0f, q = 0.0f;
    for (int pos = 0; pos < 4; ++pos)
      for (int j = 0; j < 12; ++j) {
        float v = hB[pos * DIM + tid * 12 + j];
        s += v; q += v * v;
      }
    float mu = s * (1.0f / 48.0f);
    gmu[tid] = mu;
    gis[tid] = rsqrtf(q * (1.0f / 48.0f) - mu * mu + 1e-6f);
  }
  __syncthreads();

  // GN1 affine + swish + residual
  for (int o = tid; o < 480; o += 256) {
    int ch = o % DIM, g = ch / 12;
    float v = (hB[o] - gmu[g]) * gis[g] * g1s[ch] + g1b[ch];
    out[bI * 480 + o] = xs[o] + swishf(v);
  }
}

// ---------------------------------------------------------------------------
extern "C" void kernel_launch(void* const* d_in, const int* in_sizes, int n_in,
                              void* d_out, int out_size, void* d_ws, size_t ws_size,
                              hipStream_t stream) {
  const float* x       = (const float*)d_in[0];   // (128,2,2,120)
  const float* temb    = (const float*)d_in[1];   // (128,512)
  const float* qparams = (const float*)d_in[2];   // (26,72)
  const float* c0k     = (const float*)d_in[3];   // (3,3,81,81)
  const float* c0b     = (const float*)d_in[4];
  const float* c1k     = (const float*)d_in[5];
  const float* c1b     = (const float*)d_in[6];
  const float* g0s     = (const float*)d_in[7];
  const float* g0b     = (const float*)d_in[8];
  const float* g1s     = (const float*)d_in[9];
  const float* g1b     = (const float*)d_in[10];
  const float* tdw     = (const float*)d_in[11];  // (512,240)
  const float* tdb     = (const float*)d_in[12];  // (240,)
  float* out = (float*)d_out;

  float* ws  = (float*)d_ws;
  float* h1  = ws;                 // 128*4*39 = 19968 floats
  float* k0s = ws + 19968;         // 59049 floats
  float* k1s = ws + 79017;         // 59049 floats
  float* tml = ws + 138066;        // 128*240 = 30720 floats
  _Float16* swA = (_Float16*)(ws + 168800);   // 65536 halves, 16B-aligned
  _Float16* bT  = swA + 65536;                // 122880 halves (~1.05 MB total)

  quantum_kernel<<<dim3(B_SZ, QC), 128, 0, stream>>>(x, qparams, h1);
  wstd_kernel<<<2 * CONV_C, 256, 0, stream>>>(c0k, c1k, k0s, k1s);
  prep_mlp_kernel<<<480, 256, 0, stream>>>(temb, tdw, swA, bT);
  time_mlp_wmma<<<dim3(15, 8), 32, 0, stream>>>(swA, bT, tdb, tml);
  fused_block_kernel<<<B_SZ, 256, 0, stream>>>(x, h1, tml, k0s, c0b, k1s, c1b,
                                               g0s, g0b, g1s, g1b, out);
}
